// GridNetBlock_23347442221391
// MI455X (gfx1250) — matrix operand order repaired and checked by reading, not compiled, not run on hardware
//
#include <hip/hip_runtime.h>
#include <cstdint>
#include <cstddef>

// ---------------------------------------------------------------------------
// Types
// ---------------------------------------------------------------------------
typedef __bf16 bf16_t;
typedef __attribute__((ext_vector_type(16))) __bf16 v16bf;
typedef __attribute__((ext_vector_type(8)))  __bf16 v8bf;
typedef __attribute__((ext_vector_type(8)))  float  v8f;

// Model constants
#define LLW  100
#define SS   611         // T + L - 1
#define SPAD 640         // padded S

__device__ inline v8f zero_v8f() { v8f z;
#pragma unroll
  for (int i = 0; i < 8; ++i) z[i] = 0.0f; return z; }

__device__ inline v16bf zero_v16() { v16bf z;
#pragma unroll
  for (int i = 0; i < 16; ++i) z[i] = (bf16_t)0.0f; return z; }

// Pack 4 floats -> 4 bf16 and store as one 8-byte LDS write.
__device__ inline void store4_bf16(bf16_t* dst, float4 v) {
  union { bf16_t h[4]; uint2 u; } pk;
  pk.h[0] = (bf16_t)v.x; pk.h[1] = (bf16_t)v.y;
  pk.h[2] = (bf16_t)v.z; pk.h[3] = (bf16_t)v.w;
  *reinterpret_cast<uint2*>(dst) = pk.u;
}

// A fragment for v_wmma_f32_16x16x32_bf16:
//  lane m = lane&15 is the row; per-lane halves are K = k0..k0+7 and k0+16..k0+23,
//  k0 = 8*(lane>>4) relative to the 32-wide K slab. Row data is K-contiguous.
__device__ inline v16bf load_a_frag(const bf16_t* row, int k0) {
  union { v16bf v; v8bf h[2]; } u;
  u.h[0] = *reinterpret_cast<const v8bf*>(row + k0);
  u.h[1] = *reinterpret_cast<const v8bf*>(row + k0 + 16);
  return u.v;
}

// B fragment: column n = lane&15, halves K = 8h..8h+7 and 16+8h..16+8h+7 (h = lane>>4,
// k0 = 16*h passed in); B stored as W[N][K] row-major (B[k][n] = W[n][k]).
__device__ inline v16bf load_b_frag(const bf16_t* wrow, int k0) {
  union { v16bf v; v8bf h[2]; } u;
  u.h[0] = *reinterpret_cast<const v8bf*>(wrow + (k0 >> 1));
  u.h[1] = *reinterpret_cast<const v8bf*>(wrow + (k0 >> 1) + 16);
  return u.v;
}

__device__ inline v8f wmma_bf16(v16bf a, v16bf b, v8f c) {
  return __builtin_amdgcn_wmma_f32_16x16x32_bf16(false, a, false, b, (short)0, c, false, false);
}

__device__ inline float sigmoidf_(float x) { return 1.0f / (1.0f + __expf(-x)); }

// ---------------------------------------------------------------------------
// Weight conversion kernels (f32 -> bf16, with layout permutes)
// ---------------------------------------------------------------------------
__global__ void k_w_simple(const float* __restrict__ src, bf16_t* __restrict__ dst, int n) {
  int i = blockIdx.x * 256 + threadIdx.x;
  if (i < n) dst[i] = (bf16_t)src[i];
}

// Wc_mat[o][k*C + c] = W_conv[o][c][k]    ([128][512] bf16)
__global__ void k_w_conv(const float* __restrict__ Wc, bf16_t* __restrict__ dst) {
  int i = blockIdx.x * 256 + threadIdx.x;
  if (i < 128 * 512) {
    int o = i >> 9, rem = i & 511, k = rem >> 7, c = rem & 127;
    dst[i] = (bf16_t)Wc[((o << 7) + c) * 4 + k];
  }
}

// Wd_mat[k*C + o][cin] = W_deconv[cin][o][k]    ([512][512] bf16)
__global__ void k_w_deconv(const float* __restrict__ Wd, bf16_t* __restrict__ dst) {
  int i = blockIdx.x * 256 + threadIdx.x;
  if (i < 512 * 512) {
    int row = i >> 9, cin = i & 511;
    int k = row >> 7, o = row & 127;
    dst[i] = (bf16_t)Wd[((cin << 7) + o) * 4 + k];
  }
}

// Wcat[g][0:128]=Wih[g], Wcat[g][128:384]=Whh[g]; bias = bih + bhh
__global__ void k_w_cat(const float* __restrict__ Wih, const float* __restrict__ Whh,
                        const float* __restrict__ bih, const float* __restrict__ bhh,
                        bf16_t* __restrict__ W, float* __restrict__ bias) {
  int i = blockIdx.x * 256 + threadIdx.x;
  if (i < 1024) bias[i] = bih[i] + bhh[i];
  if (i < 1024 * 384) {
    int g = i / 384, j = i % 384;
    W[i] = (bf16_t)(j < 128 ? Wih[g * 128 + j] : Whh[g * 256 + (j - 128)]);
  }
}

// ---------------------------------------------------------------------------
// Generic bf16-WMMA GEMM:  out[M,N] = act(A[M,K] @ W[N,K]^T + bias) (+ residual)
//   block tile 128x64, 8 waves (4x2), wave tile 32x32 (2x2 WMMA accumulators)
//   A staged f32->bf16 in LDS (float4 loads); B fragments straight from global.
//   A_MODE: 0 = row-major(lda), 1 = gather from Vo[BH][T][QF*VD]
//   OUT_MODE: 0 = row-major(ldo), 1 = head-split [B,NH,T,QF*head_d]
// ---------------------------------------------------------------------------
#define GBM 128
#define GBN 64
#define GBK 32

template <int A_MODE, int OUT_MODE, bool PRELU, bool RESID>
__global__ __launch_bounds__(256) void k_gemm(
    const float* __restrict__ A, int lda,
    const bf16_t* __restrict__ W, int K, int N,
    const float* __restrict__ bias, int bias_mod,
    const float* __restrict__ alpha_p,
    const float* __restrict__ residual,   // row-major [M][ldo]
    float* __restrict__ out, int ldo, int head_d)
{
  __shared__ __align__(32) bf16_t As[GBM][GBK + 8];   // row stride 40 halves (80B)
  const int tid  = threadIdx.x;
  const int lane = tid & 31, wave = tid >> 5;
  const int wm = wave >> 1, wn = wave & 1;
  const int l16 = lane & 15, lh = lane >> 4;
  const int bM = blockIdx.x * GBM, bN = blockIdx.y * GBN;
  float alpha = 0.0f;
  if constexpr (PRELU) alpha = alpha_p[0];

  v8f acc[2][2];
#pragma unroll
  for (int a = 0; a < 2; ++a)
#pragma unroll
    for (int b = 0; b < 2; ++b) acc[a][b] = zero_v8f();

  // per-thread staging coordinates: 1024 float4 units per 128x32 tile, 4 each
  const int sm  = tid >> 3;          // row   (two rows per 16 threads... m = idx>>3)
  const int sk4 = (tid & 7) * 4;     // k within slab, float4 aligned

  for (int kk = 0; kk < K; kk += GBK) {
#pragma unroll
    for (int i = 0; i < 4; ++i) {
      int m = sm + i * 32;
      int row = bM + m, kg = kk + sk4;
      float4 v;
      if constexpr (A_MODE == 0) {
        v = *reinterpret_cast<const float4*>(A + (size_t)row * lda + kg);
      } else {
        int bb = row >> 15, rem = row & 32767, t = rem >> 6, qf = rem & 63;
        int nh = kg >> 5, vv = kg & 31;   // 4-wide chunk never crosses head_d=32
        v = *reinterpret_cast<const float4*>(
            A + (((size_t)(bb * 4 + nh) * 512 + t) * 2048) + qf * 32 + vv);
      }
      store4_bf16(&As[m][sk4], v);
    }
    if constexpr (A_MODE == 0) {
      if (kk + GBK < K)
        __builtin_prefetch(A + (size_t)(bM + sm) * lda + kk + GBK + sk4, 0, 1);
    }
    __syncthreads();

    const int ak0 = 8 * lh;
    const int bk0 = 16 * lh;
    v16bf af[2], bfr[2];
#pragma unroll
    for (int mt = 0; mt < 2; ++mt)
      af[mt] = load_a_frag(&As[wm * 32 + mt * 16 + l16][0], ak0);
#pragma unroll
    for (int nt = 0; nt < 2; ++nt) {
      int n = bN + wn * 32 + nt * 16 + l16;
      bfr[nt] = (n < N) ? load_b_frag(W + (size_t)n * K + kk, bk0) : zero_v16();
    }
#pragma unroll
    for (int mt = 0; mt < 2; ++mt)
#pragma unroll
      for (int nt = 0; nt < 2; ++nt)
        acc[mt][nt] = wmma_bf16(af[mt], bfr[nt], acc[mt][nt]);
    __syncthreads();
  }

  // epilogue
#pragma unroll
  for (int mt = 0; mt < 2; ++mt) {
#pragma unroll
    for (int nt = 0; nt < 2; ++nt) {
      int n = bN + wn * 32 + nt * 16 + l16;
      if (n >= N) continue;
#pragma unroll
      for (int r = 0; r < 8; ++r) {
        int m = bM + wm * 32 + mt * 16 + r + 8 * lh;
        float v = acc[mt][nt][r];
        v += bias[n % bias_mod];
        if constexpr (PRELU) v = (v >= 0.0f) ? v : alpha * v;
        if constexpr (RESID) v += residual[(size_t)m * ldo + n];
        size_t dst;
        if constexpr (OUT_MODE == 0) {
          dst = (size_t)m * ldo + n;
        } else {
          int bb = m >> 15, rem = m & 32767, t = rem >> 6, qf = rem & 63;
          int nh = n / head_d, e = n % head_d;
          dst = (((size_t)(bb * 4 + nh) * 512 + t) * (size_t)(64 * head_d)) + qf * head_d + e;
        }
        out[dst] = v;
      }
    }
  }
}

// ---------------------------------------------------------------------------
// LayerNorm over last dim (one block per row), optional residual added after.
// ---------------------------------------------------------------------------
__global__ __launch_bounds__(256) void k_layernorm(
    const float* __restrict__ in, const float* __restrict__ gamma,
    const float* __restrict__ beta, const float* __restrict__ residual,
    float* __restrict__ out, int dim)
{
  const int row = blockIdx.x, tid = threadIdx.x;
  const float* x = in + (size_t)row * dim;
  __shared__ float red[256];
  float s = 0.0f;
  for (int i = tid; i < dim; i += 256) s += x[i];
  red[tid] = s; __syncthreads();
  for (int off = 128; off > 0; off >>= 1) { if (tid < off) red[tid] += red[tid + off]; __syncthreads(); }
  float mean = red[0] / (float)dim; __syncthreads();
  float s2 = 0.0f;
  for (int i = tid; i < dim; i += 256) { float d = x[i] - mean; s2 += d * d; }
  red[tid] = s2; __syncthreads();
  for (int off = 128; off > 0; off >>= 1) { if (tid < off) red[tid] += red[tid + off]; __syncthreads(); }
  float inv = rsqrtf(red[0] / (float)dim + 1e-5f);
  for (int i = tid; i < dim; i += 256) {
    float v = (x[i] - mean) * inv * gamma[i] + beta[i];
    if (residual) v += residual[(size_t)row * dim + i];
    out[(size_t)row * dim + i] = v;
  }
}

// ---------------------------------------------------------------------------
// Persistent LSTM: 16 sequences per workgroup, state in LDS, gates via WMMA.
//   MODE 0: intra forward  (x = z_ln [N, Q2, C], y -> y_cat[.., 0:256])
//   MODE 1: intra backward (reversed scan,       y -> y_cat[.., 256:512])
//   MODE 2: inter causal   (x = z2 [B,T,QF,C],   y -> y_i [B,T,QF,H])
// Wcat: [1024][384] bf16 = [Wih | Whh]; bias = bih + bhh (f32, 1024)
// ---------------------------------------------------------------------------
template <int MODE>
__global__ __launch_bounds__(256) void k_lstm(
    const float* __restrict__ xin, const bf16_t* __restrict__ Wcat,
    const float* __restrict__ bias,
    const float* __restrict__ h0, const float* __restrict__ c0,
    float* __restrict__ yout)
{
  __shared__ __align__(32) bf16_t a_buf[16][392];   // [x_t | h], K=384, stride 392
  __shared__ float gates[16][1024];
  __shared__ float hst[16][256];
  __shared__ float cst[16][256];
  const int tid = threadIdx.x, lane = tid & 31, wave = tid >> 5;
  const int l16 = lane & 15, lh = lane >> 4;
  const int seq0 = blockIdx.x * 16;
  const int steps = (MODE == 2) ? 512 : 16;

  for (int i = tid; i < 16 * 256; i += 256) {
    int s = i >> 8, h = i & 255;
    float hv = 0.0f, cv = 0.0f;
    if constexpr (MODE == 2) {
      hv = h0[(size_t)(seq0 + s) * 256 + h];
      cv = c0[(size_t)(seq0 + s) * 256 + h];
    }
    hst[s][h] = hv; cst[s][h] = cv;
  }
  __syncthreads();

  for (int q = 0; q < steps; ++q) {
    const int rq = (MODE == 1) ? (15 - q) : q;
    // stage x_t (16x128, 512 float4, 2 per thread)
#pragma unroll
    for (int i = 0; i < 2; ++i) {
      int idx = tid + i * 256;
      int s = idx >> 5, c4 = (idx & 31) * 4;
      size_t src;
      if constexpr (MODE == 2) {
        int j = seq0 + s; int b = j >> 6, qf = j & 63;
        src = (((size_t)b * 512 + rq) * 64 + qf) * 128 + c4;
      } else {
        src = ((size_t)(seq0 + s) * 16 + rq) * 128 + c4;
      }
      store4_bf16(&a_buf[s][c4], *reinterpret_cast<const float4*>(xin + src));
    }
    // stage h (16x256, 1024 float4, 4 per thread)
#pragma unroll
    for (int i = 0; i < 4; ++i) {
      int idx = tid + i * 256;
      int s = idx >> 6, h4 = (idx & 63) * 4;
      store4_bf16(&a_buf[s][128 + h4], *reinterpret_cast<const float4*>(&hst[s][h4]));
    }
    __syncthreads();

    // gates[16,1024] = A[16,384] @ Wcat^T ; wave w covers cols [w*128, w*128+128)
    v8f acc[8];
#pragma unroll
    for (int j = 0; j < 8; ++j) acc[j] = zero_v8f();
    for (int kk = 0; kk < 384; kk += 32) {
      v16bf af = load_a_frag(&a_buf[l16][0], kk + 8 * lh);
#pragma unroll
      for (int j = 0; j < 8; ++j) {
        int n = wave * 128 + j * 16 + l16;
        v16bf bfr = load_b_frag(Wcat + (size_t)n * 384 + kk, 16 * lh);
        acc[j] = wmma_bf16(af, bfr, acc[j]);
      }
    }
    const int gtype = wave >> 1;   // 0:i 1:f 2:g 3:o (cols 128-aligned per wave)
#pragma unroll
    for (int j = 0; j < 8; ++j) {
      int n = wave * 128 + j * 16 + l16;
#pragma unroll
      for (int r = 0; r < 8; ++r) {
        int s = r + 8 * lh;
        float v = acc[j][r] + bias[n];
        v = (gtype == 2) ? tanhf(v) : sigmoidf_(v);
        gates[s][n] = v;
      }
    }
    __syncthreads();

    // c/h update + output
    for (int i = tid; i < 16 * 256; i += 256) {
      int s = i >> 8, h = i & 255;
      float iv = gates[s][h], fv = gates[s][256 + h];
      float gv = gates[s][512 + h], ov = gates[s][768 + h];
      float c = fv * cst[s][h] + iv * gv;
      float hh = ov * tanhf(c);
      cst[s][h] = c; hst[s][h] = hh;
      size_t dst;
      if constexpr (MODE == 2) {
        int j = seq0 + s; int b = j >> 6, qf = j & 63;
        dst = (((size_t)b * 512 + rq) * 64 + qf) * 256 + h;
      } else {
        dst = ((size_t)(seq0 + s) * 16 + rq) * 512 + (MODE == 1 ? 256 : 0) + h;
      }
      yout[dst] = hh;
    }
    __syncthreads();
  }
}

// ---------------------------------------------------------------------------
// Kf[bh][s][k] (bf16, S padded to 640): rows 0..98 = K_buf, 99..610 = Km, rest 0
// ---------------------------------------------------------------------------
__global__ void k_build_kf(const float* __restrict__ Kbuf, const float* __restrict__ Km,
                           bf16_t* __restrict__ Kf) {
  int i = blockIdx.x * 256 + threadIdx.x;                 // 8*640*512 / 4
  if (i >= 8 * SPAD * 512 / 4) return;
  int k4 = (i & 127) * 4; int r = i >> 7; int s = r % SPAD; int bh = r / SPAD;
  float4 v = make_float4(0.f, 0.f, 0.f, 0.f);
  if (s < 99)       v = *reinterpret_cast<const float4*>(Kbuf + ((size_t)bh * 99 + s) * 512 + k4);
  else if (s < SS)  v = *reinterpret_cast<const float4*>(Km + ((size_t)bh * 512 + (s - 99)) * 512 + k4);
  store4_bf16(Kf + (size_t)i * 4, v);
}

// VfT[bh][d][s] (bf16, transposed so attn@V B-fragments are contiguous)
__global__ void k_build_vft(const float* __restrict__ Vbuf, const float* __restrict__ Vm,
                            bf16_t* __restrict__ VfT) {
  int i = blockIdx.x * 256 + threadIdx.x;                 // 8*2048*640
  if (i >= 8 * 2048 * SPAD) return;
  int s = i % SPAD; int r = i / SPAD; int d = r & 2047; int bh = r >> 11;
  float v = 0.0f;
  if (s < 99)       v = Vbuf[((size_t)bh * 99 + s) * 2048 + d];
  else if (s < SS)  v = Vm[((size_t)bh * 512 + (s - 99)) * 2048 + d];
  VfT[i] = (bf16_t)v;
}

// ---------------------------------------------------------------------------
// Banded causal attention. Block = (bh, 16-query tile). Band: s in [t, t+L).
// ---------------------------------------------------------------------------
__global__ __launch_bounds__(256) void k_attn(
    const float* __restrict__ Qm, const bf16_t* __restrict__ Kf,
    const bf16_t* __restrict__ VfT, float* __restrict__ Vo)
{
  __shared__ __align__(32) bf16_t q_tile[16][520];   // K=512, stride 520 halves
  __shared__ float sc[16][128];
  __shared__ __align__(32) bf16_t at[16][144];       // K=128, stride 144 halves
  __shared__ float red[16][16];
  const int tid = threadIdx.x, lane = tid & 31, wave = tid >> 5;
  const int l16 = lane & 15, lh = lane >> 4;
  const int bh = blockIdx.y, t0 = blockIdx.x * 16;

  // stage Q tile (16x512, 2048 float4, 8 per thread)
#pragma unroll
  for (int i = 0; i < 8; ++i) {
    int idx = tid + i * 256;
    int r = idx >> 7, k4 = (idx & 127) * 4;
    float4 v = *reinterpret_cast<const float4*>(Qm + ((size_t)bh * 512 + t0 + r) * 512 + k4);
    store4_bf16(&q_tile[r][k4], v);
  }
  __syncthreads();

  // scores[16,128]: wave w owns s_local tile w*16..w*16+15
  v8f accs = zero_v8f();
  {
    const int sl = wave * 16 + l16;
    const bf16_t* krow = Kf + ((size_t)bh * SPAD + t0 + sl) * 512;
    for (int kk = 0; kk < 512; kk += 32) {
      v16bf af  = load_a_frag(&q_tile[l16][0], kk + 8 * lh);
      v16bf bfr = load_b_frag(krow + kk, 16 * lh);
      accs = wmma_bf16(af, bfr, accs);
    }
  }
  const float scale = 0.044194173824159216f;   // 1/sqrt(QF*E)
#pragma unroll
  for (int r = 0; r < 8; ++r)
    sc[r + 8 * lh][wave * 16 + l16] = accs[r] * scale;
  __syncthreads();

  // softmax with band mask: 16 groups of 16 threads; each thread 8 cols
  {
    int r = tid >> 4, ln = tid & 15;
    float vals[8]; int valid[8];
    float mx = -3.4e38f;
#pragma unroll
    for (int i = 0; i < 8; ++i) {
      int j = ln * 8 + i;
      int ok = (j >= r) && (j < r + LLW) && (t0 + j < SS);
      valid[i] = ok;
      vals[i] = sc[r][j];
      if (ok && vals[i] > mx) mx = vals[i];
    }
    red[r][ln] = mx; __syncthreads();
    if (ln == 0) { float m2 = red[r][0]; for (int i = 1; i < 16; ++i) m2 = fmaxf(m2, red[r][i]); red[r][0] = m2; }
    __syncthreads();
    mx = red[r][0]; __syncthreads();
    float s = 0.0f;
#pragma unroll
    for (int i = 0; i < 8; ++i) { vals[i] = valid[i] ? __expf(vals[i] - mx) : 0.0f; s += vals[i]; }
    red[r][ln] = s; __syncthreads();
    if (ln == 0) { float s2 = 0.0f; for (int i = 0; i < 16; ++i) s2 += red[r][i]; red[r][0] = s2; }
    __syncthreads();
    float invs = 1.0f / red[r][0];
#pragma unroll
    for (int i = 0; i < 8; ++i) at[r][ln * 8 + i] = (bf16_t)(vals[i] * invs);
  }
  __syncthreads();

  // Vo[16, 2048] = attn[16,128] @ Vf ; wave w owns d-slice [w*256, w*256+256)
  for (int jj = 0; jj < 4; ++jj) {
    v8f acc2[4];
#pragma unroll
    for (int j = 0; j < 4; ++j) acc2[j] = zero_v8f();
    for (int kk = 0; kk < 128; kk += 32) {
      v16bf af = load_a_frag(&at[l16][0], kk + 8 * lh);
#pragma unroll
      for (int j = 0; j < 4; ++j) {
        int d = wave * 256 + jj * 64 + j * 16 + l16;
        v16bf bfr = load_b_frag(VfT + ((size_t)bh * 2048 + d) * SPAD + t0 + kk, 16 * lh);
        acc2[j] = wmma_bf16(af, bfr, acc2[j]);
      }
    }
#pragma unroll
    for (int j = 0; j < 4; ++j) {
      int d = wave * 256 + jj * 64 + j * 16 + l16;
#pragma unroll
      for (int r = 0; r < 8; ++r) {
        int row = r + 8 * lh;
        Vo[((size_t)bh * 512 + t0 + row) * 2048 + d] = acc2[j][r];
      }
    }
  }
}

// ---------------------------------------------------------------------------
// Host: arena layout + launch sequence
// ---------------------------------------------------------------------------
#define MB (size_t)(1 << 20)

extern "C" void kernel_launch(void* const* d_in, const int* in_sizes, int n_in,
                              void* d_out, int out_size, void* d_ws, size_t ws_size,
                              hipStream_t stream) {
  (void)in_sizes; (void)n_in; (void)out_size; (void)ws_size;
  const float* x       = (const float*)d_in[0];
  const float* W_conv  = (const float*)d_in[1];
  const float* b_conv  = (const float*)d_in[2];
  const float* a_act   = (const float*)d_in[3];
  const float* g_norm  = (const float*)d_in[4];
  const float* b_norm  = (const float*)d_in[5];
  const float* Wih_f   = (const float*)d_in[6];
  const float* Whh_f   = (const float*)d_in[7];
  const float* bih_f   = (const float*)d_in[8];
  const float* bhh_f   = (const float*)d_in[9];
  const float* Wih_b   = (const float*)d_in[10];
  const float* Whh_b   = (const float*)d_in[11];
  const float* bih_b   = (const float*)d_in[12];
  const float* bhh_b   = (const float*)d_in[13];
  const float* Wih_i   = (const float*)d_in[14];
  const float* Whh_i   = (const float*)d_in[15];
  const float* bih_i   = (const float*)d_in[16];
  const float* bhh_i   = (const float*)d_in[17];
  const float* W_dec   = (const float*)d_in[18];
  const float* b_dec   = (const float*)d_in[19];
  const float* g_in    = (const float*)d_in[20];
  const float* b_in    = (const float*)d_in[21];
  const float* W_il    = (const float*)d_in[22];
  const float* b_il    = (const float*)d_in[23];
  const float* Wq      = (const float*)d_in[24];
  const float* bq      = (const float*)d_in[25];
  const float* aq      = (const float*)d_in[26];
  const float* gq      = (const float*)d_in[27];
  const float* bgq     = (const float*)d_in[28];
  const float* Wk      = (const float*)d_in[29];
  const float* bk      = (const float*)d_in[30];
  const float* ak      = (const float*)d_in[31];
  const float* gk      = (const float*)d_in[32];
  const float* bgk     = (const float*)d_in[33];
  const float* Wv      = (const float*)d_in[34];
  const float* bv      = (const float*)d_in[35];
  const float* av      = (const float*)d_in[36];
  const float* gv      = (const float*)d_in[37];
  const float* bgv     = (const float*)d_in[38];
  const float* Wp      = (const float*)d_in[39];
  const float* bp      = (const float*)d_in[40];
  const float* ap      = (const float*)d_in[41];
  const float* gp      = (const float*)d_in[42];
  const float* bgp     = (const float*)d_in[43];
  const float* h0      = (const float*)d_in[44];
  const float* c0      = (const float*)d_in[45];
  const float* K_buf   = (const float*)d_in[46];
  const float* V_buf   = (const float*)d_in[47];
  float* out           = (float*)d_out;

  char* ws = (char*)d_ws;
  size_t off = 0;
  auto alloc = [&](size_t bytes) -> char* {
    char* p = ws + off;
    off = (off + bytes + 255) & ~(size_t)255;
    return p;
  };
  // persistent bf16 weights + biases (~3.2 MB)
  bf16_t* wc    = (bf16_t*)alloc(128 * 512 * 2);
  bf16_t* wd    = (bf16_t*)alloc(512 * 512 * 2);
  bf16_t* wcatf = (bf16_t*)alloc(1024 * 384 * 2);
  bf16_t* wcatb = (bf16_t*)alloc(1024 * 384 * 2);
  bf16_t* wcati = (bf16_t*)alloc(1024 * 384 * 2);
  bf16_t* wilb  = (bf16_t*)alloc(128 * 256 * 2);
  bf16_t* wqb   = (bf16_t*)alloc(32 * 128 * 2);
  bf16_t* wkb   = (bf16_t*)alloc(32 * 128 * 2);
  bf16_t* wvb   = (bf16_t*)alloc(128 * 128 * 2);
  bf16_t* wpb   = (bf16_t*)alloc(128 * 128 * 2);
  float* bcatf  = (float*)alloc(1024 * 4);
  float* bcatb  = (float*)alloc(1024 * 4);
  float* bcati  = (float*)alloc(1024 * 4);

  // big reusable span (144 MB), phase-overlapped
  off = (off + MB - 1) & ~(MB - 1);
  char* span = ws + off; off += 144 * MB;
  // Phase I
  float* z_pre = (float*)(span + 0 * MB);    //  8 MB  [16384,128]
  float* z_ln  = (float*)(span + 8 * MB);    //  8 MB
  float* y_cat = (float*)(span + 16 * MB);   // 32 MB  [16384,512]
  float* x1    = (float*)(span + 48 * MB);   // 32 MB  [B,T,QF,C]
  float* y_i   = (float*)(span + 80 * MB);   // 64 MB  [B,T,QF,H]
  float* z2    = (float*)(span + 0 * MB);    // 32 MB  (aliases z_pre..y_cat; live 5-6)
  // Phase II (Phase-I buffers dead by first use here)
  float*  Qm_pre = (float*)(span + 0 * MB);   //  8 MB [8,512,512]
  float*  Qm     = (float*)(span + 8 * MB);   //  8 MB
  float*  Km_pre = (float*)(span + 16 * MB);  //  8 MB
  float*  Km     = (float*)(span + 24 * MB);  //  8 MB
  float*  Vm_pre = (float*)(span + 32 * MB);  // 32 MB [8,512,2048]
  float*  Vm     = (float*)(span + 64 * MB);  // 32 MB
  bf16_t* Kf     = (bf16_t*)(span + 96 * MB); //  5 MB [8,640,512] bf16
  bf16_t* VfT    = (bf16_t*)(span + 101 * MB);// 20 MB [8,2048,640] bf16
  float*  Vo     = Vm_pre;                    // 32 MB (Vm_pre dead after LN)
  float*  ytmp   = Vm;                        // 32 MB (Vm dead after vft build)

  float* out_t = (float*)alloc(32 * MB);      // [B,T,QF,C], live to the end

  // ---- weight conversions ----
  k_w_conv  <<<(128 * 512 + 255) / 256, 256, 0, stream>>>(W_conv, wc);
  k_w_deconv<<<(512 * 512 + 255) / 256, 256, 0, stream>>>(W_dec, wd);
  k_w_cat   <<<(1024 * 384 + 255) / 256, 256, 0, stream>>>(Wih_f, Whh_f, bih_f, bhh_f, wcatf, bcatf);
  k_w_cat   <<<(1024 * 384 + 255) / 256, 256, 0, stream>>>(Wih_b, Whh_b, bih_b, bhh_b, wcatb, bcatb);
  k_w_cat   <<<(1024 * 384 + 255) / 256, 256, 0, stream>>>(Wih_i, Whh_i, bih_i, bhh_i, wcati, bcati);
  k_w_simple<<<(128 * 256 + 255) / 256, 256, 0, stream>>>(W_il, wilb, 128 * 256);
  k_w_simple<<<(32 * 128 + 255) / 256, 256, 0, stream>>>(Wq, wqb, 32 * 128);
  k_w_simple<<<(32 * 128 + 255) / 256, 256, 0, stream>>>(Wk, wkb, 32 * 128);
  k_w_simple<<<(128 * 128 + 255) / 256, 256, 0, stream>>>(Wv, wvb, 128 * 128);
  k_w_simple<<<(128 * 128 + 255) / 256, 256, 0, stream>>>(Wp, wpb, 128 * 128);

  // ---- stage 1: conv (as GEMM over [16384, 512]) + PReLU, then LN(C) ----
  k_gemm<0, 0, true, false><<<dim3(16384 / 128, 2), 256, 0, stream>>>(
      x, 512, wc, 512, 128, b_conv, 128, a_act, nullptr, z_pre, 128, 0);
  k_layernorm<<<16384, 256, 0, stream>>>(z_pre, g_norm, b_norm, nullptr, z_ln, 128);

  // ---- stage 2: intra biLSTM over Q2=16 steps ----
  k_lstm<0><<<64, 256, 0, stream>>>(z_ln, wcatf, bcatf, nullptr, nullptr, y_cat);
  k_lstm<1><<<64, 256, 0, stream>>>(z_ln, wcatb, bcatb, nullptr, nullptr, y_cat);

  // ---- stage 3: deconv GEMM + residual x -> x1 ----
  k_gemm<0, 0, false, true><<<dim3(16384 / 128, 8), 256, 0, stream>>>(
      y_cat, 512, wd, 512, 512, b_dec, 128, nullptr, x, x1, 512, 0);

  // ---- stage 4: LN(g_in) then inter causal LSTM over T=512 ----
  k_layernorm<<<65536, 256, 0, stream>>>(x1, g_in, b_in, nullptr, z2, 128);
  k_lstm<2><<<8, 256, 0, stream>>>(z2, wcati, bcati, h0, c0, y_i);

  // ---- stage 5: W_il projection + residual x1 -> out_t ----
  k_gemm<0, 0, false, true><<<dim3(65536 / 128, 2), 256, 0, stream>>>(
      y_i, 256, wilb, 256, 128, b_il, 128, nullptr, x1, out_t, 128, 0);

  // ---- stage 6: Q/K/V head projections (+PReLU), head-split store, LN ----
  k_gemm<0, 1, true, false><<<dim3(65536 / 128, 1), 256, 0, stream>>>(
      out_t, 128, wqb, 128, 32, bq, 32, aq, nullptr, Qm_pre, 0, 8);
  k_gemm<0, 1, true, false><<<dim3(65536 / 128, 1), 256, 0, stream>>>(
      out_t, 128, wkb, 128, 32, bk, 32, ak, nullptr, Km_pre, 0, 8);
  k_gemm<0, 1, true, false><<<dim3(65536 / 128, 2), 256, 0, stream>>>(
      out_t, 128, wvb, 128, 128, bv, 128, av, nullptr, Vm_pre, 0, 32);
  k_layernorm<<<4096, 256, 0, stream>>>(Qm_pre, gq, bgq, nullptr, Qm, 512);
  k_layernorm<<<4096, 256, 0, stream>>>(Km_pre, gk, bgk, nullptr, Km, 512);
  k_layernorm<<<4096, 256, 0, stream>>>(Vm_pre, gv, bgv, nullptr, Vm, 2048);

  // ---- stage 7: K/V cache concat (bf16), V transposed ----
  k_build_kf <<<(8 * SPAD * 512 / 4 + 255) / 256, 256, 0, stream>>>(K_buf, Km, Kf);
  k_build_vft<<<(8 * 2048 * SPAD + 255) / 256, 256, 0, stream>>>(V_buf, Vm, VfT);

  // ---- stage 8: banded causal attention ----
  k_attn<<<dim3(32, 8), 256, 0, stream>>>(Qm, Kf, VfT, Vo);

  // ---- stage 9: output projection (gather from Vo) + PReLU ----
  k_gemm<1, 0, true, false><<<dim3(65536 / 128, 2), 256, 0, stream>>>(
      Vo, 0, wpb, 128, 128, bp, 128, ap, nullptr, ytmp, 128, 0);

  // ---- stage 10: LN over QF*C=8192 + residual out_t -> final output ----
  k_layernorm<<<1024, 256, 0, stream>>>(ytmp, gp, bgp, out_t, out, 8192);
}
